// EGConvLayer_18708877542146
// MI455X (gfx1250) — compile-verified, matrix operand order checked by reading, not run on hardware
//
#include <hip/hip_runtime.h>
#include <hip/hip_bf16.h>

#define NN 50000
#define EE 800000
#define FF 128
#define HH 128
#define DD 3

typedef __bf16 bf16_t;
typedef __bf16 v16bf __attribute__((ext_vector_type(16)));
typedef __bf16 v8bf  __attribute__((ext_vector_type(8)));
typedef float  v8f   __attribute__((ext_vector_type(8)));

// ---------------------------------------------------------------------------
// WMMA helpers (CDNA5: v_wmma_f32_16x16x32_bf16, D = A*B + C, f32 accum)
// ---------------------------------------------------------------------------
__device__ __forceinline__ v8f wmma_bf16(v16bf a, v16bf b, v8f c) {
  return __builtin_amdgcn_wmma_f32_16x16x32_bf16(
      /*neg_a=*/false, a, /*neg_b=*/false, b,
      /*c_mod=*/(short)0, c, /*reuse_a=*/false, /*reuse_b=*/false);
}

// B fragments are pre-swizzled into LDS so each lane reads its 16 bf16
// (32 contiguous bytes -> two ds_load_b128) per (kc, nTile) fragment.
__device__ __forceinline__ v16bf load_b_frag(const bf16_t* frag, int kc, int nt, int lane) {
  return *(const v16bf*)(frag + ((((kc << 3) + nt) << 5) + lane) * 16);
}

// Stage a row-major [nKc*32, 128] f32 weight matrix into bf16 fragment-major LDS.
// Fragment element i of lane L maps to k = kc*32 + (i<8?0:16) + (L>=16?8:0) + i%8,
// n = nt*16 + L%16  (matches the ISA 16-bit A/B per-lane layout).
__device__ __forceinline__ void stage_frags(const float* __restrict__ W, bf16_t* frag,
                                            int nKc, int tid, int nthreads) {
  const int total = nKc << 12; // nKc * 8 tiles * 32 lanes * 16 elems
  for (int li = tid; li < total; li += nthreads) {
    int elem = li & 15;
    int lane = (li >> 4) & 31;
    int nt   = (li >> 9) & 7;
    int kc   = li >> 12;
    int k = (kc << 5) + ((elem & 8) << 1) + ((lane & 16) >> 1) + (elem & 7);
    int n = (nt << 4) + (lane & 15);
    frag[li] = (bf16_t)W[k * 128 + n];
  }
}

// Build an A fragment from a global f32 row: lane holds row M = lane%16,
// contiguous K runs [kb, kb+7] and [kb+16, kb+23] (kb already includes the
// lanes>=16 "+8" shift).
__device__ __forceinline__ v16bf make_a_from_global(const float* __restrict__ rp, int kb) {
  float4 f0 = *(const float4*)(rp + kb);
  float4 f1 = *(const float4*)(rp + kb + 4);
  float4 f2 = *(const float4*)(rp + kb + 16);
  float4 f3 = *(const float4*)(rp + kb + 20);
  v16bf a;
  a[0]  = (bf16_t)f0.x; a[1]  = (bf16_t)f0.y; a[2]  = (bf16_t)f0.z; a[3]  = (bf16_t)f0.w;
  a[4]  = (bf16_t)f1.x; a[5]  = (bf16_t)f1.y; a[6]  = (bf16_t)f1.z; a[7]  = (bf16_t)f1.w;
  a[8]  = (bf16_t)f2.x; a[9]  = (bf16_t)f2.y; a[10] = (bf16_t)f2.z; a[11] = (bf16_t)f2.w;
  a[12] = (bf16_t)f3.x; a[13] = (bf16_t)f3.y; a[14] = (bf16_t)f3.z; a[15] = (bf16_t)f3.w;
  return a;
}

// Build an A fragment from a bf16 row-major [16,128] LDS tile.
__device__ __forceinline__ v16bf make_a_from_lds(const bf16_t* tile, int m, int kb) {
  const bf16_t* p = tile + (m << 7) + kb;
  v8bf lo = *(const v8bf*)(p);
  v8bf hi = *(const v8bf*)(p + 16);
  return __builtin_shufflevector(lo, hi, 0,1,2,3,4,5,6,7,8,9,10,11,12,13,14,15);
}

// ---------------------------------------------------------------------------
// Kernel 0: zero the scatter-add workspace (m_i [N,H] + delta_x [N,3])
// ---------------------------------------------------------------------------
__global__ void egnn_zero_ws(float* __restrict__ ws, int n) {
  int i = blockIdx.x * blockDim.x + threadIdx.x;
  if (i < n) ws[i] = 0.0f;
}

// ---------------------------------------------------------------------------
// Kernel 1: fused edge MLP + scatter.  256 thr = 8 waves, 128 edges / block.
// LDS layout (dynamic, bytes):
//   [0      ) We1 fragments  (8 kc x 8 nt)      65536
//   [65536  ) We2 fragments  (4 kc x 8 nt)      32768
//   [98304  ) m1 staging     (8 waves x 16x128) 32768
//   [131072 ) We1 row 256 (f32 x128)              512
//   [131584 ) be1                                  512
//   [132096 ) be2                                  512
//   [132608 ) Wx                                   512
//   [133120 ) sqdist  (8x16 f32)                   512
//   [133632 ) dist    (8x16x3 f32)                1536
//   [135168 ) dstIdx  (8x16 int)                   512
//   [135680 ) wv      (8x16 f32)                   512   -> total 136192
// ---------------------------------------------------------------------------
#define EDGE_LDS_BYTES 136192

__global__ void __launch_bounds__(256)
egnn_edge(const float* __restrict__ node, const float* __restrict__ coord,
          const int* __restrict__ eidx,
          const float* __restrict__ We1, const float* __restrict__ be1,
          const float* __restrict__ We2, const float* __restrict__ be2,
          const float* __restrict__ Wx,  const float* __restrict__ bx,
          float* __restrict__ mi_ws, float* __restrict__ dx_ws) {
  extern __shared__ char smem[];
  bf16_t* we1f    = (bf16_t*)(smem);
  bf16_t* we2f    = (bf16_t*)(smem + 65536);
  bf16_t* m1      = (bf16_t*)(smem + 98304);
  float*  we1last = (float*)(smem + 131072);
  float*  sbe1    = (float*)(smem + 131584);
  float*  sbe2    = (float*)(smem + 132096);
  float*  swx     = (float*)(smem + 132608);
  float*  ssq     = (float*)(smem + 133120);
  float*  sdist   = (float*)(smem + 133632);
  int*    sdst    = (int*)  (smem + 135168);
  float*  swv     = (float*)(smem + 135680);

  const int tid = threadIdx.x;
  stage_frags(We1, we1f, 8, tid, 256);
  stage_frags(We2, we2f, 4, tid, 256);
  if (tid < 128) {
    we1last[tid] = We1[256 * 128 + tid];
    sbe1[tid] = be1[tid];
    sbe2[tid] = be2[tid];
    swx[tid]  = Wx[tid];
  }

  const int lane = tid & 31;
  const int wave = tid >> 5;
  const int m    = lane & 15;         // edge row within this wave's 16-edge tile
  const int sel8 = (lane & 16) >> 1;  // 0 or 8 (K shift / row-half shift)

  const int e  = blockIdx.x * 128 + wave * 16 + m;  // E % 128 == 0, no guard
  const int si = eidx[e];        // src
  const int di = eidx[EE + e];   // dst

  const float* __restrict__ rowi = node + (size_t)di * FF;
  const float* __restrict__ rowj = node + (size_t)si * FF;
  __builtin_prefetch(rowi, 0, 0);   // global_prefetch_b8
  __builtin_prefetch(rowj, 0, 0);

  if (lane < 16) {
    float d0 = coord[di * 3 + 0] - coord[si * 3 + 0];
    float d1 = coord[di * 3 + 1] - coord[si * 3 + 1];
    float d2 = coord[di * 3 + 2] - coord[si * 3 + 2];
    int idx = wave * 16 + m;
    sdist[idx * 3 + 0] = d0;
    sdist[idx * 3 + 1] = d1;
    sdist[idx * 3 + 2] = d2;
    ssq[idx]  = d0 * d0 + d1 * d1 + d2 * d2;
    sdst[idx] = di;
    swv[idx]  = 0.0f;
  }
  __syncthreads();

  // ---- layer 1: relu(e_in @ We1 + be1), e_in = [h_i | h_j | sqdist] -------
  const v8f vzero = {0.f, 0.f, 0.f, 0.f, 0.f, 0.f, 0.f, 0.f};
  v8f acc[8];
#pragma unroll
  for (int nt = 0; nt < 8; ++nt) acc[nt] = vzero;

#pragma unroll
  for (int kc = 0; kc < 8; ++kc) {
    const float* rp = (kc < 4) ? rowi : rowj;   // cols 0..127 = h_i, 128..255 = h_j
    int kb = ((kc & 3) << 5) + sel8;
    v16bf a = make_a_from_global(rp, kb);
#pragma unroll
    for (int nt = 0; nt < 8; ++nt) {
      v16bf b = load_b_frag(we1f, kc, nt, lane);
      acc[nt] = wmma_bf16(a, b, acc[nt]);
    }
  }

  // rank-1 update for the sqdist column (K=256) + bias + ReLU, stage to LDS
  bf16_t* m1w = m1 + wave * (16 * 128);
#pragma unroll
  for (int nt = 0; nt < 8; ++nt) {
    int n = (nt << 4) + m;
    float wlast = we1last[n];
    float bias  = sbe1[n];
#pragma unroll
    for (int v = 0; v < 8; ++v) {
      int row = v + sel8;
      float x = acc[nt][v] + ssq[wave * 16 + row] * wlast + bias;
      x = fmaxf(x, 0.0f);
      m1w[row * 128 + n] = (bf16_t)x;
    }
  }
  __syncthreads();

  // ---- layer 2: m_ij = m1 @ We2 + be2; scatter m_i and coordinate update --
  float wp[8];
#pragma unroll
  for (int v = 0; v < 8; ++v) wp[v] = 0.0f;

#pragma unroll
  for (int nt = 0; nt < 8; ++nt) {
    v8f acc2 = vzero;
#pragma unroll
    for (int kc = 0; kc < 4; ++kc) {
      v16bf a2 = make_a_from_lds(m1w, m, (kc << 5) + sel8);
      v16bf b2 = load_b_frag(we2f, kc, nt, lane);
      acc2 = wmma_bf16(a2, b2, acc2);
    }
    int n = (nt << 4) + m;
    float bias = sbe2[n];
    float wxn  = swx[n];
#pragma unroll
    for (int v = 0; v < 8; ++v) {
      int row = v + sel8;
      float val = acc2[v] + bias;            // m_ij[row][n]
      int dn = sdst[wave * 16 + row];
      atomicAdd(&mi_ws[(size_t)dn * HH + n], val);   // segment_sum(m_ij, dst)
      wp[v] += val * wxn;                    // partial of m_ij @ Wx
    }
  }
#pragma unroll
  for (int v = 0; v < 8; ++v)
    atomicAdd(&swv[wave * 16 + v + sel8], wp[v]);    // reduce over n-lanes
  __syncthreads();

  if (lane < 16) {
    int idx = wave * 16 + m;
    float w = swv[idx] + bx[0];              // w_ij
    int dn = sdst[idx];
    atomicAdd(&dx_ws[dn * 3 + 0], sdist[idx * 3 + 0] * w);
    atomicAdd(&dx_ws[dn * 3 + 1], sdist[idx * 3 + 1] * w);
    atomicAdd(&dx_ws[dn * 3 + 2], sdist[idx * 3 + 2] * w);
  }
}

// ---------------------------------------------------------------------------
// Kernel 2: node MLP  h_new = relu([node|m_i] @ Wh1 + bh1) @ Wh2 + bh2
// 128 rows / block, same WMMA machinery, direct stores.
// ---------------------------------------------------------------------------
#define NODE_LDS_BYTES 132608

__global__ void __launch_bounds__(256)
egnn_node(const float* __restrict__ node, const float* __restrict__ mi_ws,
          const float* __restrict__ Wh1, const float* __restrict__ bh1,
          const float* __restrict__ Wh2, const float* __restrict__ bh2,
          float* __restrict__ hout) {
  extern __shared__ char smem[];
  bf16_t* wh1f = (bf16_t*)(smem);
  bf16_t* wh2f = (bf16_t*)(smem + 65536);
  bf16_t* m1   = (bf16_t*)(smem + 98304);
  float*  sbh1 = (float*)(smem + 131072);
  float*  sbh2 = (float*)(smem + 131584);

  const int tid = threadIdx.x;
  stage_frags(Wh1, wh1f, 8, tid, 256);
  stage_frags(Wh2, wh2f, 4, tid, 256);
  if (tid < 128) { sbh1[tid] = bh1[tid]; sbh2[tid] = bh2[tid]; }
  __syncthreads();

  const int lane = tid & 31;
  const int wave = tid >> 5;
  const int m    = lane & 15;
  const int sel8 = (lane & 16) >> 1;

  const int rowBase = blockIdx.x * 128 + wave * 16;
  int r  = rowBase + m;
  int ri = (r < NN) ? r : (NN - 1);   // clamp gather row (stores are guarded)

  const float* __restrict__ rowh = node  + (size_t)ri * FF;
  const float* __restrict__ rowm = mi_ws + (size_t)ri * HH;

  const v8f vzero = {0.f, 0.f, 0.f, 0.f, 0.f, 0.f, 0.f, 0.f};
  v8f acc[8];
#pragma unroll
  for (int nt = 0; nt < 8; ++nt) acc[nt] = vzero;

#pragma unroll
  for (int kc = 0; kc < 8; ++kc) {
    const float* rp = (kc < 4) ? rowh : rowm;  // cols 0..127 = node, 128..255 = m_i
    int kb = ((kc & 3) << 5) + sel8;
    v16bf a = make_a_from_global(rp, kb);
#pragma unroll
    for (int nt = 0; nt < 8; ++nt) {
      v16bf b = load_b_frag(wh1f, kc, nt, lane);
      acc[nt] = wmma_bf16(a, b, acc[nt]);
    }
  }

  bf16_t* m1w = m1 + wave * (16 * 128);
#pragma unroll
  for (int nt = 0; nt < 8; ++nt) {
    int n = (nt << 4) + m;
    float bias = sbh1[n];
#pragma unroll
    for (int v = 0; v < 8; ++v) {
      int row = v + sel8;
      float x = fmaxf(acc[nt][v] + bias, 0.0f);
      m1w[row * 128 + n] = (bf16_t)x;
    }
  }
  __syncthreads();

#pragma unroll
  for (int nt = 0; nt < 8; ++nt) {
    v8f acc2 = vzero;
#pragma unroll
    for (int kc = 0; kc < 4; ++kc) {
      v16bf a2 = make_a_from_lds(m1w, m, (kc << 5) + sel8);
      v16bf b2 = load_b_frag(wh2f, kc, nt, lane);
      acc2 = wmma_bf16(a2, b2, acc2);
    }
    int n = (nt << 4) + m;
    float bias = sbh2[n];
#pragma unroll
    for (int v = 0; v < 8; ++v) {
      int g = rowBase + v + sel8;
      if (g < NN) hout[(size_t)g * FF + n] = acc2[v] + bias;
    }
  }
}

// ---------------------------------------------------------------------------
// Kernel 3: x_new = coordinate + delta_x / (N - 1)
// ---------------------------------------------------------------------------
__global__ void egnn_x(const float* __restrict__ coord, const float* __restrict__ dx_ws,
                       float* __restrict__ xout) {
  int i = blockIdx.x * blockDim.x + threadIdx.x;
  if (i < NN * DD) xout[i] = coord[i] + dx_ws[i] * (1.0f / (float)(NN - 1));
}

// ---------------------------------------------------------------------------
extern "C" void kernel_launch(void* const* d_in, const int* in_sizes, int n_in,
                              void* d_out, int out_size, void* d_ws, size_t ws_size,
                              hipStream_t stream) {
  const float* node  = (const float*)d_in[0];
  const float* coord = (const float*)d_in[1];
  const int*   eidx  = (const int*)  d_in[2];
  const float* We1   = (const float*)d_in[3];
  const float* be1   = (const float*)d_in[4];
  const float* We2   = (const float*)d_in[5];
  const float* be2   = (const float*)d_in[6];
  const float* Wx    = (const float*)d_in[7];
  const float* bx    = (const float*)d_in[8];
  const float* Wh1   = (const float*)d_in[9];
  const float* bh1   = (const float*)d_in[10];
  const float* Wh2   = (const float*)d_in[11];
  const float* bh2   = (const float*)d_in[12];

  float* hout  = (float*)d_out;                       // [N, F]
  float* xout  = (float*)d_out + (size_t)NN * FF;     // [N, D]
  float* mi_ws = (float*)d_ws;                        // [N, H]
  float* dx_ws = mi_ws + (size_t)NN * HH;             // [N, D]

  const int zn = NN * HH + NN * DD;
  egnn_zero_ws<<<(zn + 255) / 256, 256, 0, stream>>>(mi_ws, zn);

  egnn_edge<<<EE / 128, 256, EDGE_LDS_BYTES, stream>>>(
      node, coord, eidx, We1, be1, We2, be2, Wx, bx, mi_ws, dx_ws);

  egnn_node<<<(NN + 127) / 128, 256, NODE_LDS_BYTES, stream>>>(
      node, mi_ws, Wh1, bh1, Wh2, bh2, hout);

  egnn_x<<<(NN * DD + 255) / 256, 256, 0, stream>>>(coord, dx_ws, xout);
}
